// EnsembleFraudDetector_39676907880720
// MI455X (gfx1250) — matrix operand order repaired and checked by reading, not compiled, not run on hardware
//
#include <hip/hip_runtime.h>
#include <hip/hip_bf16.h>
#include <math.h>

// ---------------------------------------------------------------------------
// Problem constants (from the reference)
// ---------------------------------------------------------------------------
#define NN   100000
#define EE_  240000
#define INF_ 165
#define HID_ 64
#define OUT_ 2
#define HEADS_ 8

// ---------------------------------------------------------------------------
// Types for the CDNA5 WMMA builtins (gfx1250, wave32)
// ---------------------------------------------------------------------------
typedef __attribute__((ext_vector_type(16))) __bf16          v16bf;
typedef __attribute__((ext_vector_type(16))) unsigned short  v16u;
typedef __attribute__((ext_vector_type(8)))  unsigned short  v8u;
typedef __attribute__((ext_vector_type(8)))  float           v8f;
typedef __attribute__((ext_vector_type(4)))  unsigned        u32x4;
typedef __attribute__((ext_vector_type(8)))  unsigned        u32x8;

#define GEMM_BIAS 1
#define GEMM_ACC  2
#define GEMM_RELU 4
#define GEMM_ELU  8

static __host__ __device__ inline int ks_pad32(int k) { return (k + 31) & ~31; }

// ---------------------------------------------------------------------------
// Small device helpers
// ---------------------------------------------------------------------------
__device__ __forceinline__ unsigned short f2bf(float f) {
  unsigned u = __float_as_uint(f);
  unsigned r = u + 0x7fffu + ((u >> 16) & 1u);   // round-to-nearest-even
  return (unsigned short)(r >> 16);
}
// order-preserving encoding so atomicMax(uint) implements float max
__device__ __forceinline__ unsigned enc_f(float f) {
  unsigned u = __float_as_uint(f);
  return (u & 0x80000000u) ? ~u : (u | 0x80000000u);
}
__device__ __forceinline__ float dec_f(unsigned u) {
  return __uint_as_float((u & 0x80000000u) ? (u & 0x7fffffffu) : ~u);
}

// ---------------------------------------------------------------------------
// GEMM: C[M,N] (f32) = A[M,Kpad] (bf16, zero-padded) @ W^T[N,Kpad] (bf16)
//
// block = 128 threads (4 waves); each wave owns a 16x16 C tile.
// grid  = (M/16, ceil(N/64)).
//
// Data path (the MI455X-native one):
//  * B panel (64 x Kpad bf16 of pre-transposed weights) is DMA'd into LDS by
//    the Tensor Data Mover (tensor_load_to_lds + s_wait_tensorcnt), then each
//    lane's B fragment is two 8-bf16 runs 16 apart in LDS  -> 2 x 16B ds loads.
//  * A fragments are two contiguous 16B global loads per lane (row-major
//    16-bit data => GLOBAL_LOAD_B128 per ISA 10.9), thanks to the padded
//    32-element K stride.
//  * v_wmma_f32_16x16x32_bf16 accumulates in f32.
// ---------------------------------------------------------------------------
__global__ __launch_bounds__(128)
void gemm_bf16_wmma(const unsigned short* __restrict__ A,   // M x Ks
                    const unsigned short* __restrict__ WT,  // N x Ks (transposed weights)
                    const float* __restrict__ bias,
                    float* __restrict__ C,
                    int M, int Ks, int N, int flags)
{
  extern __shared__ unsigned short panel[];   // 64 x Ks bf16

  const int tid  = threadIdx.x;
  const int wave = tid >> 5;
  const int lane = tid & 31;
  const int m0   = blockIdx.x * 16;
  const int n0b  = blockIdx.y * 64;
  const int cols = (N - n0b) < 64 ? (N - n0b) : 64;

  // ---- TDM: stream the W^T panel for this block into LDS (one wave issues) --
  if (wave == 0) {
    unsigned long long ga = (unsigned long long)(const void*)(WT + (size_t)n0b * Ks);
    unsigned lds_off = (unsigned)(size_t)(void*)panel;   // low 32 bits = LDS byte offset
    u32x4 g0;
    g0[0] = 1u;                                          // count=1 (valid descriptor)
    g0[1] = lds_off;                                     // lds_addr
    g0[2] = (unsigned)ga;                                // global_addr[31:0]
    g0[3] = (unsigned)((ga >> 32) & 0x01FFFFFFu) | (2u << 30);  // addr[56:32] | type=2
    u32x8 g1;
    g1[0] = (1u << 16);                                  // data_size = 2 bytes
    g1[1] = ((unsigned)Ks & 0xFFFFu) << 16;              // tensor_dim0[15:0]
    g1[2] = (((unsigned)Ks >> 16) & 0xFFFFu)             // tensor_dim0[31:16]
          | (((unsigned)cols & 0xFFFFu) << 16);          // tensor_dim1[15:0]
    g1[3] = (((unsigned)Ks & 0xFFFFu) << 16);            // tensor_dim1[31:16]=0 | tile_dim0
    g1[4] = ((unsigned)cols & 0xFFFFu);                  // tile_dim1 (tile_dim2 = 0)
    g1[5] = (unsigned)Ks;                                // tensor_dim0_stride[31:0]
    g1[6] = 0u;                                          // stride hi / dim1_stride lo
    g1[7] = 0u;                                          // dim1_stride hi (unused, 2D tile)
    asm volatile("tensor_load_to_lds %0, %1" :: "s"(g0), "s"(g1) : "memory");
  }
  __builtin_amdgcn_s_wait_tensorcnt(0);
  __syncthreads();

  // ---- per-lane fragment coordinates (ISA 7.12.2, bf16, wave32) ------------
  const int mrow = lane & 15;
  const int kA   = (lane >> 4) << 3;            // lanes 16-31: +8 within each 16-span
  const int ncol = lane & 15;
  const int kB   = (lane >> 4) << 3;            // lanes 16-31: +8 within each 16-span
  const int lrow = (wave << 4) | ncol;          // row inside the 64-row LDS panel

  int gm = m0 + mrow; if (gm > M - 1) gm = M - 1;
  const unsigned short* __restrict__ arow = A + (size_t)gm * Ks + kA;
  const unsigned short* __restrict__ brow = panel + (size_t)lrow * Ks + kB;

  v8f acc = {};
  for (int k0 = 0; k0 < Ks; k0 += 32) {
    // A fragment: two contiguous 16B global loads (global_load_b128)
    v8u a0 = *(const v8u*)(arow + k0);
    v8u a1 = *(const v8u*)(arow + k0 + 16);
    // B fragment: two 8-bf16 runs from the LDS panel (2 x ds_load_b128)
    v8u b0 = *(const v8u*)(brow + k0);
    v8u b1 = *(const v8u*)(brow + k0 + 16);
    if (k0 + 32 < Ks) {
      __builtin_prefetch(arow + k0 + 32, 0, 0);   // global_prefetch next A tile
    }
    v16u au = __builtin_shufflevector(a0, a1, 0,1,2,3,4,5,6,7,8,9,10,11,12,13,14,15);
    v16u bu = __builtin_shufflevector(b0, b1, 0,1,2,3,4,5,6,7,8,9,10,11,12,13,14,15);
    acc = __builtin_amdgcn_wmma_f32_16x16x32_bf16(
        /*neg_a=*/false, __builtin_bit_cast(v16bf, au),
        /*neg_b=*/false, __builtin_bit_cast(v16bf, bu),
        /*c_mod=*/(short)0, acc, /*reuse_a=*/false, /*reuse_b=*/false);
  }

  // ---- epilogue: C layout = lanes 0-15 N=lane, vgpr r -> M=r (16-31: M=8+r)
  const int gn = n0b + (wave << 4) + ncol;
  if (gn < N) {
    const int rbase = (lane >> 4) << 3;
#pragma unroll
    for (int r = 0; r < 8; ++r) {
      int gmr = m0 + rbase + r;
      if (gmr < M) {
        float v = acc[r];
        if (flags & GEMM_BIAS) v += bias[gn];
        float* p = C + (size_t)gmr * N + gn;
        if (flags & GEMM_ACC)  v += *p;
        if (flags & GEMM_RELU) v = fmaxf(v, 0.0f);
        if (flags & GEMM_ELU)  v = (v > 0.0f) ? v : (__expf(v) - 1.0f);
        *p = v;
      }
    }
  }
}

// ---------------------------------------------------------------------------
// Elementwise / graph kernels
// ---------------------------------------------------------------------------
// activations: f32 [M,K] -> bf16 [M,Ks] zero-padded
__global__ void cast_pad_kernel(const float* __restrict__ in,
                                unsigned short* __restrict__ out,
                                int M, int K, int Ks) {
  long long i = (long long)blockIdx.x * blockDim.x + threadIdx.x;
  if (i >= (long long)M * Ks) return;
  int m = (int)(i / Ks), k = (int)(i % Ks);
  out[i] = (k < K) ? f2bf(in[(size_t)m * K + k]) : (unsigned short)0;
}

// weights: f32 [K,N] -> bf16 W^T [N,Ks] zero-padded
__global__ void cast_transpose_pad_kernel(const float* __restrict__ W,
                                          unsigned short* __restrict__ WT,
                                          int K, int N, int Ks) {
  long long i = (long long)blockIdx.x * blockDim.x + threadIdx.x;
  if (i >= (long long)N * Ks) return;
  int n = (int)(i / Ks), k = (int)(i % Ks);
  WT[i] = (k < K) ? f2bf(W[(size_t)k * N + n]) : (unsigned short)0;
}

// int64 edges -> int32, with N self-loops appended
__global__ void edge_prep_kernel(const long long* __restrict__ ei,
                                 int E, int N,
                                 int* __restrict__ src32, int* __restrict__ dst32) {
  long long i = (long long)blockIdx.x * blockDim.x + threadIdx.x;
  int EE = E + N;
  if (i >= EE) return;
  if (i < E) { src32[i] = (int)ei[i]; dst32[i] = (int)ei[(long long)E + i]; }
  else       { src32[i] = (int)(i - E); dst32[i] = (int)(i - E); }
}

__global__ void degree_kernel(const int* __restrict__ dst, int E,
                              float* __restrict__ deg) {
  long long i = (long long)blockIdx.x * blockDim.x + threadIdx.x;
  if (i < E) atomicAdd(&deg[dst[i]], 1.0f);
}

__global__ void scatter_add_rows_kernel(const float* __restrict__ h,
                                        const int* __restrict__ src,
                                        const int* __restrict__ dst,
                                        int E, int C, float* __restrict__ out) {
  long long i = (long long)blockIdx.x * blockDim.x + threadIdx.x;
  long long total = (long long)E * C;
  if (i >= total) return;
  int e = (int)(i / C), c = (int)(i % C);
  atomicAdd(&out[(size_t)dst[e] * C + c], h[(size_t)src[e] * C + c]);
}

__global__ void scale_rows_kernel(float* __restrict__ x, const float* __restrict__ deg,
                                  int N, int C) {
  long long i = (long long)blockIdx.x * blockDim.x + threadIdx.x;
  long long total = (long long)N * C;
  if (i >= total) return;
  x[i] /= fmaxf(deg[i / C], 1.0f);
}

// a_s[n,h] = sum_c z[n,h,c]*asrc[h,c] ; a_d likewise
__global__ void gat_node_scores_kernel(const float* __restrict__ z,
                                       const float* __restrict__ asrc,
                                       const float* __restrict__ adst,
                                       int N, int H, int C,
                                       float* __restrict__ a_s,
                                       float* __restrict__ a_d) {
  long long i = (long long)blockIdx.x * blockDim.x + threadIdx.x;
  if (i >= (long long)N * H) return;
  int n = (int)(i / H), h = (int)(i % H);
  const float* zp = z + (size_t)n * H * C + (size_t)h * C;
  float s0 = 0.f, s1 = 0.f;
  for (int c = 0; c < C; ++c) { s0 += zp[c] * asrc[h * C + c]; s1 += zp[c] * adst[h * C + c]; }
  a_s[i] = s0; a_d[i] = s1;
}

__global__ void edge_scores_leaky_kernel(const float* __restrict__ a_s,
                                         const float* __restrict__ a_d,
                                         const int* __restrict__ src,
                                         const int* __restrict__ dst,
                                         int E, int H, float slope,
                                         float* __restrict__ scores) {
  long long i = (long long)blockIdx.x * blockDim.x + threadIdx.x;
  if (i >= (long long)E * H) return;
  int e = (int)(i / H), h = (int)(i % H);
  float v = a_s[(size_t)src[e] * H + h] + a_d[(size_t)dst[e] * H + h];
  scores[i] = (v > 0.f) ? v : slope * v;
}

// transformer: logits[e,h] = dot(q[dst,h,:], k[src,h,:]) * scale
__global__ void edge_scores_dot_kernel(const float* __restrict__ q,
                                       const float* __restrict__ k,
                                       const int* __restrict__ src,
                                       const int* __restrict__ dst,
                                       int E, int H, int C, float scale,
                                       float* __restrict__ scores) {
  long long i = (long long)blockIdx.x * blockDim.x + threadIdx.x;
  if (i >= (long long)E * H) return;
  int e = (int)(i / H), h = (int)(i % H);
  const float* qp = q + (size_t)dst[e] * H * C + (size_t)h * C;
  const float* kp = k + (size_t)src[e] * H * C + (size_t)h * C;
  float s = 0.f;
  for (int c = 0; c < C; ++c) s += qp[c] * kp[c];
  scores[i] = s * scale;
}

__global__ void seg_max_kernel(const float* __restrict__ scores,
                               const int* __restrict__ dst, int E, int H,
                               unsigned* __restrict__ m_enc) {
  long long i = (long long)blockIdx.x * blockDim.x + threadIdx.x;
  if (i >= (long long)E * H) return;
  int e = (int)(i / H), h = (int)(i % H);
  atomicMax(&m_enc[(size_t)dst[e] * H + h], enc_f(scores[i]));
}

__global__ void seg_expsum_kernel(const float* __restrict__ scores,
                                  const int* __restrict__ dst, int E, int H,
                                  const unsigned* __restrict__ m_enc,
                                  float* __restrict__ ssum) {
  long long i = (long long)blockIdx.x * blockDim.x + threadIdx.x;
  if (i >= (long long)E * H) return;
  int e = (int)(i / H), h = (int)(i % H);
  size_t d = (size_t)dst[e] * H + h;
  atomicAdd(&ssum[d], __expf(scores[i] - dec_f(m_enc[d])));
}

// out[dst,h,c] += vals[src,h,c] * softmax_alpha(e,h)
__global__ void seg_aggregate_kernel(const float* __restrict__ scores,
                                     const int* __restrict__ src,
                                     const int* __restrict__ dst,
                                     int E, int H, int C,
                                     const unsigned* __restrict__ m_enc,
                                     const float* __restrict__ ssum,
                                     const float* __restrict__ vals,
                                     float* __restrict__ out) {
  long long i = (long long)blockIdx.x * blockDim.x + threadIdx.x;
  long long total = (long long)E * H * C;
  if (i >= total) return;
  int e = (int)(i / ((long long)H * C));
  int r = (int)(i % ((long long)H * C));
  int h = r / C, c = r % C;
  size_t dh = (size_t)dst[e] * H + h;
  float alpha = __expf(scores[(size_t)e * H + h] - dec_f(m_enc[dh])) /
                fmaxf(ssum[dh], 1e-16f);
  atomicAdd(&out[(size_t)dst[e] * H * C + (size_t)h * C + c],
            vals[(size_t)src[e] * H * C + (size_t)h * C + c] * alpha);
}

// x = act(x + bias?) ; act: 0 none, 1 elu, 2 relu ; bias may be null
__global__ void bias_act_kernel(float* __restrict__ x, const float* __restrict__ bias,
                                int N, int C, int act) {
  long long i = (long long)blockIdx.x * blockDim.x + threadIdx.x;
  if (i >= (long long)N * C) return;
  float v = x[i];
  if (bias) v += bias[i % C];
  if (act == 1) v = (v > 0.f) ? v : (__expf(v) - 1.f);
  else if (act == 2) v = fmaxf(v, 0.f);
  x[i] = v;
}

__global__ void combine_kernel(const float* __restrict__ s, const float* __restrict__ g,
                               const float* __restrict__ t, const float* __restrict__ ew,
                               float* __restrict__ out, long long n) {
  long long i = (long long)blockIdx.x * blockDim.x + threadIdx.x;
  if (i >= n) return;
  float w0 = ew[0], w1 = ew[1], w2 = ew[2];
  float mx = fmaxf(w0, fmaxf(w1, w2));
  float e0 = __expf(w0 - mx), e1 = __expf(w1 - mx), e2 = __expf(w2 - mx);
  float inv = 1.0f / (e0 + e1 + e2);
  out[i] = (e0 * s[i] + e1 * g[i] + e2 * t[i]) * inv;
}

// ---------------------------------------------------------------------------
// Host-side orchestration
// ---------------------------------------------------------------------------
static inline unsigned cdiv_u(long long a, long long b) { return (unsigned)((a + b - 1) / b); }

extern "C" void kernel_launch(void* const* d_in, const int* in_sizes, int n_in,
                              void* d_out, int out_size, void* d_ws, size_t ws_size,
                              hipStream_t stream) {
  (void)in_sizes; (void)n_in; (void)out_size; (void)ws_size;

  const int N = NN, E = EE_, IN = INF_, HID = HID_, H = HEADS_;
  const int EE = E + N;
  const int HC = H * HID;                 // 512
  const int KS_IN = ks_pad32(IN);         // 192
  const int KS_HC = ks_pad32(HC);         // 512
  const int KS_HID = ks_pad32(HID);       // 64

  // ---- inputs (setup_inputs dict order, params flattened in insertion order)
  const float*     x    = (const float*)d_in[0];
  const long long* eidx = (const long long*)d_in[1];
  const float* sage_Wl[3] = { (const float*)d_in[2], (const float*)d_in[3], (const float*)d_in[4] };
  const float* sage_bl[3] = { (const float*)d_in[5], (const float*)d_in[6], (const float*)d_in[7] };
  const float* sage_Wr[3] = { (const float*)d_in[8], (const float*)d_in[9], (const float*)d_in[10] };
  const float* gat1_W = (const float*)d_in[11];
  const float* gat1_as = (const float*)d_in[12];
  const float* gat1_ad = (const float*)d_in[13];
  const float* gat1_b = (const float*)d_in[14];
  const float* gat2_W = (const float*)d_in[15];
  const float* gat2_as = (const float*)d_in[16];
  const float* gat2_ad = (const float*)d_in[17];
  const float* gat2_b = (const float*)d_in[18];
  const float* gcls_W = (const float*)d_in[19];
  const float* gcls_b = (const float*)d_in[20];
  const float* t1_W[4] = { (const float*)d_in[21], (const float*)d_in[23],
                           (const float*)d_in[25], (const float*)d_in[27] };   // q,k,v,skip
  const float* t1_b[4] = { (const float*)d_in[22], (const float*)d_in[24],
                           (const float*)d_in[26], (const float*)d_in[28] };
  const float* t2_W[4] = { (const float*)d_in[29], (const float*)d_in[31],
                           (const float*)d_in[33], (const float*)d_in[35] };
  const float* t2_b[4] = { (const float*)d_in[30], (const float*)d_in[32],
                           (const float*)d_in[34], (const float*)d_in[36] };
  const float* tcls_W = (const float*)d_in[37];
  const float* tcls_b = (const float*)d_in[38];
  const float* ens_w  = (const float*)d_in[39];

  // ---- workspace bump allocator
  char* wp = (char*)d_ws;
  auto alloc = [&](size_t bytes) -> void* {
    void* p = (void*)wp;
    wp += (bytes + 255) & ~((size_t)255);
    return p;
  };

  unsigned short* xb  = (unsigned short*)alloc((size_t)N * KS_IN * 2);  // bf16 x (padded)
  unsigned short* hb  = (unsigned short*)alloc((size_t)N * KS_HC * 2);  // generic bf16 act
  unsigned short* hb2 = (unsigned short*)alloc((size_t)N * KS_HID * 2); // sage root bf16
  int*   src32 = (int*)alloc((size_t)EE * 4);
  int*   dst32 = (int*)alloc((size_t)EE * 4);
  float* deg   = (float*)alloc((size_t)N * 4);
  float* bufA  = (float*)alloc((size_t)N * HC * 4);   // agg / skip / g / t1
  float* bufB  = (float*)alloc((size_t)N * HC * 4);   // z1 / q
  float* bufC  = (float*)alloc((size_t)N * HC * 4);   // k   (reused N*64 later)
  float* bufD  = (float*)alloc((size_t)N * HC * 4);   // v   (reused N*64 later)
  float* h64a  = (float*)alloc((size_t)N * HID * 4);
  float* h64b  = (float*)alloc((size_t)N * HID * 4);
  float* sage_out = (float*)alloc((size_t)N * OUT_ * 4);
  float* gat_out  = (float*)alloc((size_t)N * OUT_ * 4);
  float* tr_out   = (float*)alloc((size_t)N * OUT_ * 4);
  float* a_s   = (float*)alloc((size_t)N * H * 4);
  float* a_d   = (float*)alloc((size_t)N * H * 4);
  float* scores = (float*)alloc((size_t)EE * H * 4);
  unsigned* m_enc = (unsigned*)alloc((size_t)N * H * 4);
  float*    ssum  = (float*)alloc((size_t)N * H * 4);

  // transposed + padded bf16 weights: W[K,N] -> WT[N, ks_pad32(K)]
  auto wprep = [&](const float* src, int K, int Ncols) -> unsigned short* {
    int Ks = ks_pad32(K);
    unsigned short* dst = (unsigned short*)alloc((size_t)Ncols * Ks * 2);
    cast_transpose_pad_kernel<<<cdiv_u((long long)Ncols * Ks, 256), 256, 0, stream>>>(
        src, dst, K, Ncols, Ks);
    return dst;
  };
  unsigned short* sWl_b[3]; unsigned short* sWr_b[3];
  const int sKin[3]  = { IN, HID, HID };
  const int sKout[3] = { HID, HID, OUT_ };
  for (int l = 0; l < 3; ++l) {
    sWl_b[l] = wprep(sage_Wl[l], sKin[l], sKout[l]);
    sWr_b[l] = wprep(sage_Wr[l], sKin[l], sKout[l]);
  }
  unsigned short* g1W_b   = wprep(gat1_W, IN, HC);
  unsigned short* g2W_b   = wprep(gat2_W, HC, HID);
  unsigned short* gclsW_b = wprep(gcls_W, HID, OUT_);
  unsigned short* t1W_b[4]; unsigned short* t2W_b[4];
  for (int i = 0; i < 4; ++i) {
    t1W_b[i] = wprep(t1_W[i], IN, HC);
    t2W_b[i] = wprep(t2_W[i], HC, HID);
  }
  unsigned short* tclsW_b = wprep(tcls_W, HID, OUT_);

  auto castA = [&](const float* src, unsigned short* dst, int M, int K) {
    int Ks = ks_pad32(K);
    cast_pad_kernel<<<cdiv_u((long long)M * Ks, 256), 256, 0, stream>>>(src, dst, M, K, Ks);
  };
  auto gemm = [&](const unsigned short* A, const unsigned short* WT, const float* bias,
                  float* C, int M, int K, int Nc, int flags) {
    int Ks = ks_pad32(K);
    dim3 g(cdiv_u(M, 16), cdiv_u(Nc, 64));
    size_t shmem = (size_t)64 * Ks * 2;   // LDS B panel filled by the TDM
    gemm_bf16_wmma<<<g, 128, shmem, stream>>>(A, WT, bias, C, M, Ks, Nc, flags);
  };

  // ---- prep: edges, degree, bf16(x)
  edge_prep_kernel<<<cdiv_u(EE, 256), 256, 0, stream>>>(eidx, E, N, src32, dst32);
  hipMemsetAsync(deg, 0, (size_t)N * 4, stream);
  degree_kernel<<<cdiv_u(E, 256), 256, 0, stream>>>(dst32, E, deg);
  castA(x, xb, N, IN);

  // ============================ GraphSAGE =================================
  {
    const float* hcur = x;                 // f32 master of current layer input
    const unsigned short* hroot = xb;      // bf16 (padded) of same
    for (int l = 0; l < 3; ++l) {
      int Cin = sKin[l], Cout = sKout[l];
      float* out = (l == 0) ? h64a : (l == 1) ? h64b : sage_out;
      // mean aggregation over real edges
      hipMemsetAsync(bufA, 0, (size_t)N * Cin * 4, stream);
      scatter_add_rows_kernel<<<cdiv_u((long long)E * Cin, 256), 256, 0, stream>>>(
          hcur, src32, dst32, E, Cin, bufA);
      scale_rows_kernel<<<cdiv_u((long long)N * Cin, 256), 256, 0, stream>>>(bufA, deg, N, Cin);
      castA(bufA, hb, N, Cin);
      gemm(hb, sWl_b[l], sage_bl[l], out, N, Cin, Cout, GEMM_BIAS);
      gemm(hroot, sWr_b[l], nullptr, out, N, Cin, Cout,
           GEMM_ACC | ((l < 2) ? GEMM_RELU : 0));
      if (l < 2) {
        hcur = out;
        castA(out, hb2, N, Cout);
        hroot = hb2;
      }
    }
  }

  // ============================== GAT =====================================
  {
    // layer 1: z1 = x @ W1  (N x 512)
    gemm(xb, g1W_b, nullptr, bufB, N, IN, HC, 0);
    gat_node_scores_kernel<<<cdiv_u((long long)N * H, 256), 256, 0, stream>>>(
        bufB, gat1_as, gat1_ad, N, H, HID, a_s, a_d);
    edge_scores_leaky_kernel<<<cdiv_u((long long)EE * H, 256), 256, 0, stream>>>(
        a_s, a_d, src32, dst32, EE, H, 0.2f, scores);
    hipMemsetAsync(m_enc, 0, (size_t)N * H * 4, stream);
    seg_max_kernel<<<cdiv_u((long long)EE * H, 256), 256, 0, stream>>>(scores, dst32, EE, H, m_enc);
    hipMemsetAsync(ssum, 0, (size_t)N * H * 4, stream);
    seg_expsum_kernel<<<cdiv_u((long long)EE * H, 256), 256, 0, stream>>>(scores, dst32, EE, H, m_enc, ssum);
    hipMemsetAsync(bufA, 0, (size_t)N * HC * 4, stream);
    seg_aggregate_kernel<<<cdiv_u((long long)EE * H * HID, 256), 256, 0, stream>>>(
        scores, src32, dst32, EE, H, HID, m_enc, ssum, bufB, bufA);
    bias_act_kernel<<<cdiv_u((long long)N * HC, 256), 256, 0, stream>>>(bufA, gat1_b, N, HC, 1); // elu

    // layer 2 (H=1, C=64): z2 = g @ W2
    castA(bufA, hb, N, HC);
    gemm(hb, g2W_b, nullptr, h64a, N, HC, HID, 0);
    gat_node_scores_kernel<<<cdiv_u((long long)N, 256), 256, 0, stream>>>(
        h64a, gat2_as, gat2_ad, N, 1, HID, a_s, a_d);
    edge_scores_leaky_kernel<<<cdiv_u((long long)EE, 256), 256, 0, stream>>>(
        a_s, a_d, src32, dst32, EE, 1, 0.2f, scores);
    hipMemsetAsync(m_enc, 0, (size_t)N * 4, stream);
    seg_max_kernel<<<cdiv_u((long long)EE, 256), 256, 0, stream>>>(scores, dst32, EE, 1, m_enc);
    hipMemsetAsync(ssum, 0, (size_t)N * 4, stream);
    seg_expsum_kernel<<<cdiv_u((long long)EE, 256), 256, 0, stream>>>(scores, dst32, EE, 1, m_enc, ssum);
    hipMemsetAsync(h64b, 0, (size_t)N * HID * 4, stream);
    seg_aggregate_kernel<<<cdiv_u((long long)EE * HID, 256), 256, 0, stream>>>(
        scores, src32, dst32, EE, 1, HID, m_enc, ssum, h64a, h64b);
    bias_act_kernel<<<cdiv_u((long long)N * HID, 256), 256, 0, stream>>>(h64b, gat2_b, N, HID, 0);

    castA(h64b, hb, N, HID);
    gemm(hb, gclsW_b, gcls_b, gat_out, N, HID, OUT_, GEMM_BIAS);
  }

  // ========================= TransformerConv ==============================
  {
    // layer 1: q,k,v and skip (skip doubles as aggregation accumulator)
    gemm(xb, t1W_b[0], t1_b[0], bufB, N, IN, HC, GEMM_BIAS);   // q
    gemm(xb, t1W_b[1], t1_b[1], bufC, N, IN, HC, GEMM_BIAS);   // k
    gemm(xb, t1W_b[2], t1_b[2], bufD, N, IN, HC, GEMM_BIAS);   // v
    gemm(xb, t1W_b[3], t1_b[3], bufA, N, IN, HC, GEMM_BIAS);   // skip -> accumulator
    const float scale = 0.125f;   // 1/sqrt(64)
    edge_scores_dot_kernel<<<cdiv_u((long long)E * H, 256), 256, 0, stream>>>(
        bufB, bufC, src32, dst32, E, H, HID, scale, scores);
    hipMemsetAsync(m_enc, 0, (size_t)N * H * 4, stream);
    seg_max_kernel<<<cdiv_u((long long)E * H, 256), 256, 0, stream>>>(scores, dst32, E, H, m_enc);
    hipMemsetAsync(ssum, 0, (size_t)N * H * 4, stream);
    seg_expsum_kernel<<<cdiv_u((long long)E * H, 256), 256, 0, stream>>>(scores, dst32, E, H, m_enc, ssum);
    seg_aggregate_kernel<<<cdiv_u((long long)E * H * HID, 256), 256, 0, stream>>>(
        scores, src32, dst32, E, H, HID, m_enc, ssum, bufD, bufA);  // += onto skip
    bias_act_kernel<<<cdiv_u((long long)N * HC, 256), 256, 0, stream>>>(bufA, nullptr, N, HC, 1); // elu

    // layer 2 (H=1, C=64)
    castA(bufA, hb, N, HC);
    gemm(hb, t2W_b[0], t2_b[0], h64a, N, HC, HID, GEMM_BIAS);  // q2
    gemm(hb, t2W_b[1], t2_b[1], h64b, N, HC, HID, GEMM_BIAS);  // k2
    gemm(hb, t2W_b[2], t2_b[2], bufC, N, HC, HID, GEMM_BIAS);  // v2
    gemm(hb, t2W_b[3], t2_b[3], bufD, N, HC, HID, GEMM_BIAS);  // skip2 -> accumulator
    edge_scores_dot_kernel<<<cdiv_u((long long)E, 256), 256, 0, stream>>>(
        h64a, h64b, src32, dst32, E, 1, HID, scale, scores);
    hipMemsetAsync(m_enc, 0, (size_t)N * 4, stream);
    seg_max_kernel<<<cdiv_u((long long)E, 256), 256, 0, stream>>>(scores, dst32, E, 1, m_enc);
    hipMemsetAsync(ssum, 0, (size_t)N * 4, stream);
    seg_expsum_kernel<<<cdiv_u((long long)E, 256), 256, 0, stream>>>(scores, dst32, E, 1, m_enc, ssum);
    seg_aggregate_kernel<<<cdiv_u((long long)E * HID, 256), 256, 0, stream>>>(
        scores, src32, dst32, E, 1, HID, m_enc, ssum, bufC, bufD);  // += onto skip2

    castA(bufD, hb, N, HID);
    gemm(hb, tclsW_b, tcls_b, tr_out, N, HID, OUT_, GEMM_BIAS);
  }

  // ========================= ensemble combine =============================
  combine_kernel<<<cdiv_u((long long)N * OUT_, 256), 256, 0, stream>>>(
      sage_out, gat_out, tr_out, ens_w, (float*)d_out, (long long)N * OUT_);
}
